// GNN_90598040142035
// MI455X (gfx1250) — matrix operand order repaired and checked by reading, not compiled, not run on hardware
//
#include <hip/hip_runtime.h>
#include <hip/hip_bf16.h>
#include <math.h>

typedef __attribute__((ext_vector_type(16))) _Float16 v16h;
typedef __attribute__((ext_vector_type(8)))  float    v8f;

#define NEG_SLOPE 0.2f

// ---------- float <-> orderable-uint for atomic max over floats ----------
__device__ __forceinline__ unsigned f2ord(float f) {
    unsigned u = __float_as_uint(f);
    return (u & 0x80000000u) ? ~u : (u | 0x80000000u);
}
__device__ __forceinline__ float ord2f(unsigned o) {
    unsigned u = (o & 0x80000000u) ? (o & 0x7fffffffu) : ~o;
    return __uint_as_float(u);
}

// ---------- per-node init + layer1 scores (in_dim==1 collapse) ----------
__global__ void node1_init(const float* __restrict__ x,
                           const float* __restrict__ W1,
                           const float* __restrict__ a1s,
                           const float* __restrict__ a1d,
                           float* s1, float* d1,
                           unsigned* m1, float* sum1, float* agg1,
                           unsigned* m2, float* sum2, float* out2,
                           float* pooled, float* cnt,
                           int N, int G) {
    int i = blockIdx.x * blockDim.x + threadIdx.x;
    if (i < G) {
        cnt[i] = 0.0f;
        for (int j = 0; j < 32; ++j) pooled[(size_t)i * 32 + j] = 0.0f;
    }
    if (i >= N) return;
    float cs = 0.0f, cd = 0.0f;
    for (int j = 0; j < 64; ++j) {
        float w = W1[j];
        cs += w * a1s[j];
        cd += w * a1d[j];
    }
    float xi = x[i];
    s1[i] = xi * cs;
    d1[i] = xi * cd;
    m1[i] = 0u;  sum1[i] = 0.0f;  agg1[i] = 0.0f;
    m2[i] = 0u;  sum2[i] = 0.0f;
    for (int j = 0; j < 32; ++j) out2[(size_t)i * 32 + j] = 0.0f;
}

// ---------- edge pass A: alpha = leaky_relu(s[src]+d[dst]); segment max ----------
__global__ void edge_passA(const int* __restrict__ srcI, const int* __restrict__ dstI,
                           int E, int N,
                           const float* __restrict__ s, const float* __restrict__ d,
                           float* __restrict__ alpha, unsigned* __restrict__ mOrd) {
    int e = blockIdx.x * blockDim.x + threadIdx.x;
    if (e >= E + N) return;
    int si = (e < E) ? srcI[e] : (e - E);
    int di = (e < E) ? dstI[e] : (e - E);
    float a = s[si] + d[di];
    a = (a > 0.0f) ? a : NEG_SLOPE * a;
    alpha[e] = a;
    atomicMax(&mOrd[di], f2ord(a));
}

// ---------- edge pass B: e = exp(alpha - m[dst]); segment sum ----------
__global__ void edge_passB(const int* __restrict__ srcI, const int* __restrict__ dstI,
                           int E, int N,
                           float* __restrict__ alpha,
                           const unsigned* __restrict__ mOrd,
                           float* __restrict__ sum) {
    int e = blockIdx.x * blockDim.x + threadIdx.x;
    if (e >= E + N) return;
    int di = (e < E) ? dstI[e] : (e - E);
    float ev = __expf(alpha[e] - ord2f(mOrd[di]));
    alpha[e] = ev;
    atomicAdd(&sum[di], ev);
}

// ---------- layer1 edge pass C (scalar aggregation: in_dim==1) ----------
__global__ void edge_passC1(const int* __restrict__ srcI, const int* __restrict__ dstI,
                            int E, int N,
                            const float* __restrict__ ealpha,
                            const float* __restrict__ sum1,
                            const float* __restrict__ x,
                            float* __restrict__ agg1) {
    int e = blockIdx.x * blockDim.x + threadIdx.x;
    if (e >= E + N) return;
    int si = (e < E) ? srcI[e] : (e - E);
    int di = (e < E) ? dstI[e] : (e - E);
    float coef = ealpha[e] / sum1[di];
    atomicAdd(&agg1[di], coef * x[si]);
}

// ---------- finalize layer1: h1relu (f16) = relu(agg1[i]*W1[j] + b1[j]) ----------
__global__ void finalize1(const float* __restrict__ agg1,
                          const float* __restrict__ W1,
                          const float* __restrict__ b1,
                          _Float16* __restrict__ h1h, int N) {
    int i = blockIdx.x * blockDim.x + threadIdx.x;
    if (i >= N) return;
    float a = agg1[i];
    for (int j = 0; j < 64; ++j) {
        float h = a * W1[j] + b1[j];
        h = (h > 0.0f) ? h : 0.0f;
        h1h[(size_t)i * 64 + j] = (_Float16)h;
    }
}

// ---------- W2 f32 -> f16, stored TRANSPOSED: W2t[n*64+k] = W2[k*32+n] ----------
__global__ void convW2t(const float* __restrict__ W2, _Float16* __restrict__ W2t) {
    int i = blockIdx.x * blockDim.x + threadIdx.x;
    if (i >= 64 * 32) return;
    int k = i >> 5, n = i & 31;
    W2t[(size_t)n * 64 + k] = (_Float16)W2[(size_t)k * 32 + n];
}

// ---------- h2[N,32] = h1relu[N,64] @ W2[64,32] via v_wmma_f32_16x16x32_f16 ----------
// Block = 128 threads = 4 waves; wave w -> (row tile w>>1, col tile w&1).
// A (32x64) and Bt (32x64, N-major) staged to LDS with global_load_async_to_lds_b128.
// With Bt N-major, B-fragment reads have the same dword pattern as A-fragment
// reads -> coalescible ds_load_b128s, no half-insert packing.
__global__ void __launch_bounds__(128)
gemm2_wmma(const _Float16* __restrict__ h1h,
           const _Float16* __restrict__ W2t,
           float* __restrict__ h2, int N) {
    __shared__ __align__(16) _Float16 Atile[32 * 64];   // 4 KB
    __shared__ __align__(16) _Float16 Btile[32 * 64];   // 4 KB (N-major: [n][k])
    int t    = threadIdx.x;           // 0..127
    int lane = t & 31;
    int wave = t >> 5;                // 0..3
    int rowt = wave >> 1;             // row tile within block
    int colt = wave & 1;              // col tile within block
    int row0 = blockIdx.x * 32;

    // low 32 bits of the generic pointer == LDS byte offset (AMDGPU addrspacecast)
    unsigned ldsA = (unsigned)(size_t)&Atile[0];
    unsigned ldsB = (unsigned)(size_t)&Btile[0];

    // async copy: 256 x 16B chunks each for A (row-clamped) and Bt
    #pragma unroll
    for (int i = 0; i < 2; ++i) {
        int ch = t + 128 * i;                   // chunk id 0..255
        int r  = row0 + (ch >> 3);              // 8 chunks (128B) per A row
        if (r > N - 1) r = N - 1;
        unsigned gOffA = (unsigned)r * 128u + (unsigned)(ch & 7) * 16u;
        unsigned lOffA = ldsA + (unsigned)ch * 16u;
        asm volatile("global_load_async_to_lds_b128 %0, %1, %2"
                     :: "v"(lOffA), "v"(gOffA), "s"(h1h) : "memory");
        unsigned gOffB = (unsigned)ch * 16u;
        unsigned lOffB = ldsB + (unsigned)ch * 16u;
        asm volatile("global_load_async_to_lds_b128 %0, %1, %2"
                     :: "v"(lOffB), "v"(gOffB), "s"(W2t) : "memory");
    }
    asm volatile("s_wait_asynccnt 0" ::: "memory");
    __syncthreads();

    int m  = lane & 15;                         // A row within tile
    int kh = lane >> 4;                         // lane-half selects K group
    int nc = (lane & 15) + colt * 16;           // output column

    const unsigned* Arow = (const unsigned*)&Atile[(rowt * 16 + m) * 64];
    const unsigned* Brow = (const unsigned*)&Btile[nc * 64];
    union AV { v16h v; unsigned u[8]; };
    v8f acc = {};
    #pragma unroll
    for (int kb = 0; kb < 64; kb += 32) {
        AV a, b;
        #pragma unroll
        for (int v = 0; v < 8; ++v) {
            // A 16-bit 16x32 layout (ISA 7.12.2): VGPR v, lane-half kh
            int Ka = kb + 8 * kh + ((v < 4) ? (2 * v) : (16 + 2 * (v - 4)));
            a.u[v] = Arow[Ka >> 1];             // packed pair (Ka, Ka+1)
            // B 16-bit 32x16: lane = N column, same K order as A;
            // with Bt N-major the dword index matches A's.
            b.u[v] = Brow[Ka >> 1];
        }
        acc = __builtin_amdgcn_wmma_f32_16x16x32_f16(
                  /*neg_a=*/false, a.v, /*neg_b=*/false, b.v,
                  /*c_mod=*/(short)0, acc, /*reuse_a=*/false, /*reuse_b=*/false);
    }

    int rbase = row0 + rowt * 16 + 8 * kh;      // C/D layout: VGPR r -> M = r + 8*kh
    if (row0 + 32 <= N) {
        #pragma unroll
        for (int r = 0; r < 8; ++r)
            h2[(size_t)(rbase + r) * 32 + nc] = acc[r];
    } else {
        #pragma unroll
        for (int r = 0; r < 8; ++r)
            if (rbase + r < N) h2[(size_t)(rbase + r) * 32 + nc] = acc[r];
    }
}

// ---------- layer2 per-node scores ----------
__global__ void node2_scores(const float* __restrict__ h2,
                             const float* __restrict__ a2s,
                             const float* __restrict__ a2d,
                             float* __restrict__ s2, float* __restrict__ d2, int N) {
    int i = blockIdx.x * blockDim.x + threadIdx.x;
    if (i >= N) return;
    const float* row = h2 + (size_t)i * 32;
    float ss = 0.0f, dd = 0.0f;
    for (int j = 0; j < 32; ++j) { float h = row[j]; ss += h * a2s[j]; dd += h * a2d[j]; }
    s2[i] = ss; d2[i] = dd;
}

// ---------- layer2 edge pass C: out2[dst,:] += coef * h2[src,:] ----------
__global__ void edge_passC2(const int* __restrict__ srcI, const int* __restrict__ dstI,
                            int E, int N,
                            const float* __restrict__ ealpha,
                            const float* __restrict__ sum2,
                            const float* __restrict__ h2,
                            float* __restrict__ out2) {
    long long t = (long long)blockIdx.x * blockDim.x + threadIdx.x;
    long long tot = (long long)(E + N) * 32;
    if (t >= tot) return;
    int e = (int)(t >> 5), j = (int)(t & 31);
    int si = (e < E) ? srcI[e] : (e - E);
    int di = (e < E) ? dstI[e] : (e - E);
    const float* hrow = h2 + (size_t)si * 32;
    if (j == 0) __builtin_prefetch(hrow, 0, 1);   // global_prefetch_b8
    float coef = ealpha[e] / sum2[di];
    atomicAdd(&out2[(size_t)di * 32 + j], coef * hrow[j]);
}

// ---------- relu + bias + global mean-pool accumulation ----------
__global__ void pool_accum(const float* __restrict__ out2,
                           const float* __restrict__ b2,
                           const int* __restrict__ batch,
                           float* __restrict__ pooled, float* __restrict__ cnt, int N) {
    int i = blockIdx.x * blockDim.x + threadIdx.x;
    if (i >= N) return;
    int g = batch[i];
    for (int j = 0; j < 32; ++j) {
        float v = out2[(size_t)i * 32 + j] + b2[j];
        v = (v > 0.0f) ? v : 0.0f;
        atomicAdd(&pooled[(size_t)g * 32 + j], v);
    }
    atomicAdd(&cnt[g], 1.0f);
}

// ---------- final linear: [G,32] @ [32,2] + bl ----------
__global__ void final_linear(const float* __restrict__ pooled,
                             const float* __restrict__ cnt,
                             const float* __restrict__ Wl,
                             const float* __restrict__ bl,
                             float* __restrict__ out, int G) {
    int t = blockIdx.x * blockDim.x + threadIdx.x;
    if (t >= G * 2) return;
    int g = t >> 1, c = t & 1;
    float n = cnt[g]; n = (n > 1.0f) ? n : 1.0f;
    float acc = bl[c];
    for (int j = 0; j < 32; ++j)
        acc += (pooled[(size_t)g * 32 + j] / n) * Wl[j * 2 + c];
    out[t] = acc;
}

extern "C" void kernel_launch(void* const* d_in, const int* in_sizes, int n_in,
                              void* d_out, int out_size, void* d_ws, size_t ws_size,
                              hipStream_t stream) {
    const float* x   = (const float*)d_in[0];
    const int*   ei  = (const int*)d_in[1];
    const int*   bat = (const int*)d_in[2];
    const float* W1  = (const float*)d_in[3];
    const float* a1s = (const float*)d_in[4];
    const float* a1d = (const float*)d_in[5];
    const float* b1  = (const float*)d_in[6];
    const float* W2  = (const float*)d_in[7];
    const float* a2s = (const float*)d_in[8];
    const float* a2d = (const float*)d_in[9];
    const float* b2  = (const float*)d_in[10];
    const float* Wl  = (const float*)d_in[11];
    const float* bl  = (const float*)d_in[12];

    const int N = in_sizes[0];
    const int E = in_sizes[1] / 2;
    const int G = out_size / 2;
    const int Etot = E + N;
    const int* srcI = ei;
    const int* dstI = ei + E;

    // ---- workspace carve-out (256B aligned) ----
    char* w = (char*)d_ws;
    size_t off = 0;
    auto alloc = [&](size_t bytes) -> void* {
        void* p = w + off;
        off += (bytes + 255) & ~size_t(255);
        return p;
    };
    float*     s1    = (float*)alloc((size_t)N * 4);
    float*     d1    = (float*)alloc((size_t)N * 4);
    unsigned*  m1    = (unsigned*)alloc((size_t)N * 4);
    float*     sum1  = (float*)alloc((size_t)N * 4);
    float*     agg1  = (float*)alloc((size_t)N * 4);
    float*     alpha = (float*)alloc((size_t)Etot * 4);   // reused for both layers
    _Float16*  h1h   = (_Float16*)alloc((size_t)N * 64 * 2);
    _Float16*  W2t   = (_Float16*)alloc((size_t)64 * 32 * 2);
    float*     h2    = (float*)alloc((size_t)N * 32 * 4);
    float*     s2    = (float*)alloc((size_t)N * 4);
    float*     d2    = (float*)alloc((size_t)N * 4);
    unsigned*  m2    = (unsigned*)alloc((size_t)N * 4);
    float*     sum2  = (float*)alloc((size_t)N * 4);
    float*     out2  = (float*)alloc((size_t)N * 32 * 4);
    float*     pooled= (float*)alloc((size_t)G * 32 * 4);
    float*     cnt   = (float*)alloc((size_t)G * 4);
    (void)ws_size;

    const int T = 256;
    int gN  = (N + T - 1) / T;
    int gNG = ((N > G ? N : G) + T - 1) / T;
    int gE  = (Etot + T - 1) / T;
    long long c2tot = (long long)Etot * 32;
    int gC2 = (int)((c2tot + T - 1) / T);

    // Layer 1 (scalar-collapsed GAT)
    node1_init<<<gNG, T, 0, stream>>>(x, W1, a1s, a1d, s1, d1, m1, sum1, agg1,
                                      m2, sum2, out2, pooled, cnt, N, G);
    edge_passA<<<gE, T, 0, stream>>>(srcI, dstI, E, N, s1, d1, alpha, m1);
    edge_passB<<<gE, T, 0, stream>>>(srcI, dstI, E, N, alpha, m1, sum1);
    edge_passC1<<<gE, T, 0, stream>>>(srcI, dstI, E, N, alpha, sum1, x, agg1);
    finalize1<<<gN, T, 0, stream>>>(agg1, W1, b1, h1h, N);

    // Layer 2 feature transform via WMMA (async LDS staging, transposed B)
    convW2t<<<8, T, 0, stream>>>(W2, W2t);
    gemm2_wmma<<<(N + 31) / 32, 128, 0, stream>>>(h1h, W2t, h2, N);
    node2_scores<<<gN, T, 0, stream>>>(h2, a2s, a2d, s2, d2, N);

    // Layer 2 GAT aggregation
    edge_passA<<<gE, T, 0, stream>>>(srcI, dstI, E, N, s2, d2, alpha, m2);
    edge_passB<<<gE, T, 0, stream>>>(srcI, dstI, E, N, alpha, m2, sum2);
    edge_passC2<<<gC2, T, 0, stream>>>(srcI, dstI, E, N, alpha, sum2, h2, out2);

    // Pool + classifier
    pool_accum<<<gN, T, 0, stream>>>(out2, b2, bat, pooled, cnt, N);
    final_linear<<<(G * 2 + T - 1) / T, T, 0, stream>>>(pooled, cnt, Wl, bl,
                                                        (float*)d_out, G);
}
